// CDE_64785286693495
// MI455X (gfx1250) — matrix operand order repaired and checked by reading, not compiled, not run on hardware
//
#include <hip/hip_runtime.h>
#include <hip/hip_bf16.h>
#include <cmath>

typedef __attribute__((ext_vector_type(16))) _Float16 v16h;
typedef __attribute__((ext_vector_type(8)))  _Float16 v8h;
typedef __attribute__((ext_vector_type(8)))  float    v8f;

#define B_TOT   128
#define T_PTS   512
#define CCH     32
#define HDIM    64
#define DD      128         // D1 == D2
#define HC      2048        // H*C
#define OUTD    32
#define NSTEP   (T_PTS - 1) // 511
#define MB      16          // batch tile per workgroup (== WMMA M/N)
#define NWG     (B_TOT / MB)
#define NTHREADS 256

// padded LDS row lengths (keep 16B alignment, spread banks)
#define HP   (HDIM + 1)     // f32 state rows
#define HFP  (HDIM + 8)     // f16 rows (64 -> 72 halves = 144B)
#define DFP  (DD + 8)       // f16 rows (128 -> 136 halves = 272B)
#define CP   (CCH + 4)      // f32 dY rows (32 -> 36 floats = 144B)

// ---------------------------------------------------------------------------
// Prologue: pack W3 (f32 [DD][HC]) into f16 WMMA **A-fragments** of W3^T.
// A-layout (ISA 7.12.2, 16-bit A 16x32): lane L -> row M = n-offset = L&15;
//   element e<8  -> K = kk*32 + (L>>4)*8 + e
//   element e>=8 -> K = kk*32 + 16 + (L>>4)*8 + (e-8)
// Stored per-lane contiguous: frag*512 + lane*16 + e -> two aligned 16B loads.
// ---------------------------------------------------------------------------
__global__ void w3_pack_kernel(const float* __restrict__ W3,
                               _Float16* __restrict__ w3p) {
    int tid = blockIdx.x * blockDim.x + threadIdx.x;   // 0 .. DD*HC-1
    if (tid >= DD * HC) return;
    int f  = tid >> 9;
    int r  = tid & 511;
    int L  = r >> 4;
    int e  = r & 15;
    int j  = f >> 2, kk = f & 3;
    int half = L >> 4;
    int n  = j * 16 + (L & 15);
    int K  = kk * 32 + (e < 8 ? half * 8 + e : 16 + half * 8 + (e - 8));
    w3p[tid] = (_Float16)W3[(size_t)K * HC + n];
}

#define LOG2E 1.4426950408889634f
#define LN2   0.6931471805599453f

// softplus(x) = max(x,0) + ln2*log2(1 + 2^(-|x|*log2e)); arg of log2 in [1,2]
__device__ __forceinline__ float softplus_f(float x) {
    float q = __builtin_amdgcn_exp2f(-fabsf(x) * LOG2E);   // raw v_exp_f32
    float l = __builtin_amdgcn_logf(1.f + q);              // raw v_log_f32
    return fmaxf(x, 0.f) + LN2 * l;
}
// tanh(x) = 1 - 2/(2^(2x*log2e) + 1)
__device__ __forceinline__ float tanh_f(float x) {
    float e = __builtin_amdgcn_exp2f(x * (2.f * LOG2E));
    return 1.f - 2.f * __builtin_amdgcn_rcpf(e + 1.f);
}
__device__ __forceinline__ v16h cat8(v8h lo, v8h hi) {
    return __builtin_shufflevector(lo, hi, 0,1,2,3,4,5,6,7,8,9,10,11,12,13,14,15);
}

// ---------------------------------------------------------------------------
// Persistent scan: one WG per 16-batch tile, 8 waves, all 511 steps in-kernel.
// ---------------------------------------------------------------------------
__launch_bounds__(NTHREADS, 1)
__global__ void cde_scan_kernel(const float* __restrict__ ys,
                                const float* __restrict__ Winit,
                                const float* __restrict__ binit,
                                const float* __restrict__ W1,
                                const float* __restrict__ b1,
                                const float* __restrict__ b2,
                                const float* __restrict__ b3,
                                const float* __restrict__ Wr,
                                const float* __restrict__ br,
                                const float* __restrict__ W2raw,
                                const _Float16* __restrict__ w3p,
                                float* __restrict__ out) {
    __shared__ _Float16 sW1t[DD][HFP];    // W1^T (h rows only) [n][k], B-layout rows
    __shared__ _Float16 sW2t[DD][DFP];    // W2^T [n][k]
    __shared__ _Float16 sact1h[MB][DFP];  // softplus(L1), f16 row-major
    __shared__ _Float16 sact2h[MB][DFP];  // softplus(L2), f16 row-major
    __shared__ _Float16 szhh[MB][HFP];    // f16 mirror of current drift input h
    __shared__ float sW1r0[DD], sb1[DD], sb2[DD], sbias1[DD];
    __shared__ float sb3[HC];
    __shared__ float sz[MB][HP];          // z
    __shared__ float szh[MB][HP];         // zhat
    __shared__ float sfA[MB][HP];         // fhat ping
    __shared__ float sfB[MB][HP];         // fhat pong
    __shared__ float sdy[MB][CP];

    const int tid  = threadIdx.x;
    const int wave = tid >> 5;
    const int lane = tid & 31;
    const int ln   = lane & 15;
    const int half = lane >> 4;
    const int b0   = blockIdx.x * MB;

    // ---- stage weights into LDS ----
    for (int i = tid; i < DD * HDIM; i += NTHREADS) {
        int n = i >> 6, k = i & 63;
        sW1t[n][k] = (_Float16)W1[(size_t)(1 + k) * DD + n];
    }
    for (int i = tid; i < DD * DD; i += NTHREADS) {
        int n = i >> 7, k = i & 127;
        sW2t[n][k] = (_Float16)W2raw[(size_t)k * DD + n];
    }
    for (int i = tid; i < DD; i += NTHREADS) {
        sW1r0[i] = W1[i]; sb1[i] = b1[i]; sb2[i] = b2[i];
    }
    for (int i = tid; i < HC; i += NTHREADS) sb3[i] = b3[i];

    // ---- h0 = ys[:,0,:] @ W_init + b_init ----
    for (int i = tid; i < MB * HDIM; i += NTHREADS) {
        int b = i >> 6, hh = i & 63;
        const float* yb = ys + (size_t)(b0 + b) * T_PTS * CCH;
        float acc = binit[hh];
        #pragma unroll 8
        for (int c = 0; c < CCH; ++c) acc += yb[c] * Winit[(size_t)c * HDIM + hh];
        sz[b][hh] = acc; szh[b][hh] = acc; szhh[b][hh] = (_Float16)acc;
    }
    __syncthreads();

    // ---- drift: fo[b][h] = tanh-MLP(t, szhh) . dY[seg] ----
    auto drift = [&](float t, float (*fo)[HP], int seg) {
        for (int i = tid; i < DD; i += NTHREADS) sbias1[i] = sb1[i] + t * sW1r0[i];
        for (int i = tid; i < MB * CCH; i += NTHREADS) {
            int b = i >> 5, c = i & 31;
            const float* yb = ys + (size_t)(b0 + b) * T_PTS * CCH;
            sdy[b][c] = yb[(size_t)(seg + 1) * CCH + c] - yb[(size_t)seg * CCH + c];
        }
        __syncthreads();

        // ===== GEMM1: A = h [16x64] f16, B = W1^T tiles -> act1, softplus =====
        {
            v16h a[2];
            #pragma unroll
            for (int kk = 0; kk < 2; ++kk) {
                v8h lo = *(const v8h*)&szhh[ln][kk * 32 + half * 8];
                v8h hi = *(const v8h*)&szhh[ln][kk * 32 + 16 + half * 8];
                a[kk] = cat8(lo, hi);
            }
            int j = wave;
            v8f acc = *(const v8f*)&sbias1[j * 16 + 8 * half];   // bias as C-init
            #pragma unroll
            for (int kk = 0; kk < 2; ++kk) {
                v8h blo = *(const v8h*)&sW1t[j * 16 + ln][kk * 32 + half * 8];
                v8h bhi = *(const v8h*)&sW1t[j * 16 + ln][kk * 32 + 16 + half * 8];
                acc = __builtin_amdgcn_wmma_f32_16x16x32_f16(
                          false, a[kk], false, cat8(blo, bhi), (short)0, acc, false, false);
            }
            #pragma unroll
            for (int v = 0; v < 8; ++v)
                sact1h[v + 8 * half][j * 16 + ln] = (_Float16)softplus_f(acc[v]);
        }
        __syncthreads();

        // ===== GEMM2: A = act1 [16x128] f16, B = W2^T tiles -> act2, softplus =====
        {
            v16h a[4];
            #pragma unroll
            for (int kk = 0; kk < 4; ++kk) {
                v8h lo = *(const v8h*)&sact1h[ln][kk * 32 + half * 8];
                v8h hi = *(const v8h*)&sact1h[ln][kk * 32 + 16 + half * 8];
                a[kk] = cat8(lo, hi);
            }
            int j = wave;
            v8f acc = *(const v8f*)&sb2[j * 16 + 8 * half];
            #pragma unroll
            for (int kk = 0; kk < 4; ++kk) {
                v8h blo = *(const v8h*)&sW2t[j * 16 + ln][kk * 32 + half * 8];
                v8h bhi = *(const v8h*)&sW2t[j * 16 + ln][kk * 32 + 16 + half * 8];
                acc = __builtin_amdgcn_wmma_f32_16x16x32_f16(
                          false, a[kk], false, cat8(blo, bhi), (short)0, acc, false, false);
            }
            #pragma unroll
            for (int v = 0; v < 8; ++v)
                sact2h[v + 8 * half][j * 16 + ln] = (_Float16)softplus_f(acc[v]);
        }
        __syncthreads();

        // ===== GEMM3 (transposed): D[n][b] = W3^T-tile x L2^T, +b3, tanh, c-sum ====
        // D lands batch-in-lane (col b = ln), tile columns (h,c) in acc regs ->
        // c-contraction = 16 in-register FMAs per h + ONE shfl_xor(16) per h.
        {
            v16h bact[4];                       // B-frags: L2^T rows of act2h
            #pragma unroll
            for (int kk = 0; kk < 4; ++kk) {
                v8h lo = *(const v8h*)&sact2h[ln][kk * 32 + half * 8];
                v8h hi = *(const v8h*)&sact2h[ln][kk * 32 + 16 + half * 8];
                bact[kk] = cat8(lo, hi);
            }
            // per-lane dY regs: dyv[v] -> c = 8*half+v ; dyv[8+v] -> c = 16+8*half+v
            float dyv[16];
            *(float4*)&dyv[0]  = *(const float4*)&sdy[ln][8 * half];
            *(float4*)&dyv[4]  = *(const float4*)&sdy[ln][8 * half + 4];
            *(float4*)&dyv[8]  = *(const float4*)&sdy[ln][16 + 8 * half];
            *(float4*)&dyv[12] = *(const float4*)&sdy[ln][16 + 8 * half + 4];

            #pragma clang loop unroll_count(2)
            for (int jp = 0; jp < 8; ++jp) {    // one h = pair of 16-col tiles
                const int j0 = wave * 16 + 2 * jp;
                v8f acc0 = *(const v8f*)&sb3[j0 * 16 + 8 * half];
                #pragma unroll
                for (int kk = 0; kk < 4; ++kk) {
                    v16h aw = *(const v16h*)(w3p + ((size_t)(j0 * 4 + kk)) * 512 + lane * 16);
                    acc0 = __builtin_amdgcn_wmma_f32_16x16x32_f16(
                               false, aw, false, bact[kk], (short)0, acc0, false, false);
                }
                v8f acc1 = *(const v8f*)&sb3[(j0 + 1) * 16 + 8 * half];
                #pragma unroll
                for (int kk = 0; kk < 4; ++kk) {
                    v16h aw = *(const v16h*)(w3p + ((size_t)((j0 + 1) * 4 + kk)) * 512 + lane * 16);
                    acc1 = __builtin_amdgcn_wmma_f32_16x16x32_f16(
                               false, aw, false, bact[kk], (short)0, acc1, false, false);
                }
                float tot = 0.f;
                #pragma unroll
                for (int v = 0; v < 8; ++v)
                    tot += tanh_f(acc0[v]) * dyv[v] + tanh_f(acc1[v]) * dyv[8 + v];
                tot += __shfl_xor(tot, 16, 32);       // merge the two c-halves
                if (half == 0)
                    fo[ln][wave * 8 + jp] = tot;
            }
        }
        __syncthreads();
    };

    // fhat0 = drift(0, h0, dY[0])
    drift(0.f, sfA, 0);

    // ---- reversible Heun scan ----
    int flip = 0;
    for (int n = 0; n < NSTEP; ++n) {
        float (*fc)[HP] = flip ? sfB : sfA;
        float (*fn)[HP] = flip ? sfA : sfB;
        for (int i = tid; i < MB * HDIM; i += NTHREADS) {   // zhat1 = 2z - zhat + fhat
            int b = i >> 6, h = i & 63;
            float zh1 = 2.f * sz[b][h] - szh[b][h] + fc[b][h];
            szh[b][h] = zh1;
            szhh[b][h] = (_Float16)zh1;
        }
        __syncthreads();
        int seg = (n + 1 < T_PTS - 1) ? (n + 1) : (T_PTS - 2);
        drift((float)(n + 1), fn, seg);
        for (int i = tid; i < MB * HDIM; i += NTHREADS) {   // z += 0.5*(fhat + f1)
            int b = i >> 6, h = i & 63;
            sz[b][h] += 0.5f * (fc[b][h] + fn[b][h]);
        }
        __syncthreads();
        flip ^= 1;
    }

    // ---- readout: out = z @ Wr + br ----
    for (int i = tid; i < MB * OUTD; i += NTHREADS) {
        int b = i >> 5, o = i & 31;
        float acc = br[o];
        #pragma unroll 8
        for (int hh = 0; hh < HDIM; ++hh) acc += sz[b][hh] * Wr[(size_t)hh * OUTD + o];
        out[(size_t)(b0 + b) * OUTD + o] = acc;
    }
}

extern "C" void kernel_launch(void* const* d_in, const int* in_sizes, int n_in,
                              void* d_out, int out_size, void* d_ws, size_t ws_size,
                              hipStream_t stream) {
    const float* ys    = (const float*)d_in[0];
    const float* Winit = (const float*)d_in[1];
    const float* binit = (const float*)d_in[2];
    const float* W1    = (const float*)d_in[3];
    const float* b1    = (const float*)d_in[4];
    const float* W2    = (const float*)d_in[5];
    const float* b2    = (const float*)d_in[6];
    const float* W3    = (const float*)d_in[7];
    const float* b3    = (const float*)d_in[8];
    const float* Wr    = (const float*)d_in[9];
    const float* br    = (const float*)d_in[10];
    float* outp        = (float*)d_out;

    _Float16* w3p = (_Float16*)d_ws;   // 512 KB packed W3 A-fragments

    w3_pack_kernel<<<(DD * HC + 255) / 256, 256, 0, stream>>>(W3, w3p);
    cde_scan_kernel<<<NWG, NTHREADS, 0, stream>>>(
        ys, Winit, binit, W1, b1, b2, b3, Wr, br, W2, w3p, outp);
}
